// CMPNN_61314953118479
// MI455X (gfx1250) — matrix operand order repaired and checked by reading, not compiled, not run on hardware
//
#include <hip/hip_runtime.h>
#include <hip/hip_bf16.h>
#include <math.h>

// ---------------- problem constants (match reference) ----------------
static constexpr int HD   = 256;          // hidden
static constexpr int INC  = 128;          // in channels
static constexpr int EDIM = 64;           // edge dim
static constexpr int NG   = 1024;         // graphs
static constexpr int LP   = 48;           // atoms per graph
static constexpr int NA   = NG * LP;      // atoms  = 49152
static constexpr int NE   = 2 * NA;       // edges  = 98304

// ---------------- GEMM tiling ----------------
static constexpr int BM = 128, BN = 128, BK = 32;
static constexpr int LDA_S = 40;          // bf16 halves per LDS row (32 + pad)

typedef __attribute__((ext_vector_type(16))) __bf16 bf16x16;
typedef __attribute__((ext_vector_type(8)))  float  f32x8;

__device__ __forceinline__ unsigned short f2bfu(float f) {
  union { float f; unsigned u; } c; c.f = f;
  unsigned u = c.u + 0x7FFFu + ((c.u >> 16) & 1u);   // RNE to bf16
  return (unsigned short)(u >> 16);
}
__device__ __forceinline__ __bf16 f2bf(float f) {
  union { unsigned short s; __bf16 b; } r; r.s = f2bfu(f);
  return r.b;
}
__device__ __forceinline__ uint2 pack4bf(float4 a) {
  uint2 u;
  u.x = (unsigned)f2bfu(a.x) | ((unsigned)f2bfu(a.y) << 16);
  u.y = (unsigned)f2bfu(a.z) | ((unsigned)f2bfu(a.w) << 16);
  return u;
}

// C[M,Nn] = act(A[M,K] @ B[K,Nn] + bias), f32 memory, bf16 WMMA compute.
// Software-pipelined: tile k+1 global loads overlap tile k WMMA work.
// Requires M%128==0, Nn%128==0, K%32==0.
__global__ __launch_bounds__(256) void gemm_bf16_wmma(
    const float* __restrict__ A, const float* __restrict__ B,
    const float* __restrict__ bias, float* __restrict__ C,
    int M, int Nn, int K, int act)
{
  __shared__ __align__(16) __bf16 Alds[BM * LDA_S];
  __shared__ __align__(16) __bf16 Blds[BN * LDA_S];

  const int tid  = threadIdx.x;
  const int lane = tid & 31;
  const int w    = tid >> 5;
  const int wm   = w & 3;            // wave row group: wm*32
  const int wn   = w >> 2;           // wave col group: wn*64
  const int blockM = blockIdx.y * BM;
  const int blockN = blockIdx.x * BN;
  const int l15 = lane & 15;
  const int lhi = lane >> 4;

  f32x8 acc[2][4];
  const f32x8 fzero = {};
  #pragma unroll
  for (int mi = 0; mi < 2; ++mi)
    #pragma unroll
    for (int ni = 0; ni < 4; ++ni) acc[mi][ni] = fzero;

  float4 ra[4], rb[4];
  auto load_tile = [&](int k0) {
    #pragma unroll
    for (int j = 0; j < 4; ++j) {
      int chunk = tid + j * 256;               // 1024 float4 chunks each
      int r = chunk >> 3, kq = (chunk & 7) * 4;
      ra[j] = *(const float4*)(A + (size_t)(blockM + r) * K + (k0 + kq));
      int kk = chunk >> 5, nq = (chunk & 31) * 4;
      rb[j] = *(const float4*)(B + (size_t)(k0 + kk) * Nn + (blockN + nq));
    }
  };

  load_tile(0);
  for (int k0 = 0; k0 < K; k0 += BK) {
    __syncthreads();                           // LDS free (prev readers done)
    // convert + store current tile (A: packed b64; B: transposed [n][k])
    #pragma unroll
    for (int j = 0; j < 4; ++j) {
      int chunk = tid + j * 256;
      int r = chunk >> 3, kq = (chunk & 7) * 4;
      *(uint2*)(&Alds[r * LDA_S + kq]) = pack4bf(ra[j]);
      int kk = chunk >> 5, nq = (chunk & 31) * 4;
      Blds[(nq + 0) * LDA_S + kk] = f2bf(rb[j].x);
      Blds[(nq + 1) * LDA_S + kk] = f2bf(rb[j].y);
      Blds[(nq + 2) * LDA_S + kk] = f2bf(rb[j].z);
      Blds[(nq + 3) * LDA_S + kk] = f2bf(rb[j].w);
    }
    __syncthreads();
    // issue next tile's global loads now; they complete under the WMMA burst
    if (k0 + BK < K) load_tile(k0 + BK);
    if (k0 + 2 * BK < K) {                     // L2 hint two tiles ahead
      __builtin_prefetch(A + (size_t)(blockM + (tid >> 1)) * K + (k0 + 2 * BK) + (tid & 1) * 16, 0, 1);
      __builtin_prefetch(B + (size_t)(k0 + 2 * BK + (tid >> 3)) * Nn + blockN + (tid & 7) * 16, 0, 1);
    }

    union FB { uint4 q[2]; bf16x16 v; } fb[4];
    #pragma unroll
    for (int ni = 0; ni < 4; ++ni) {
      const __bf16* bp = &Blds[(wn * 64 + ni * 16 + l15) * LDA_S + lhi * 8];
      fb[ni].q[0] = *(const uint4*)(bp);
      fb[ni].q[1] = *(const uint4*)(bp + 16);
    }
    #pragma unroll
    for (int mi = 0; mi < 2; ++mi) {
      union FA { uint4 q[2]; bf16x16 v; } fa;
      const __bf16* ap = &Alds[(wm * 32 + mi * 16 + l15) * LDA_S + lhi * 8];
      fa.q[0] = *(const uint4*)(ap);
      fa.q[1] = *(const uint4*)(ap + 16);
      #pragma unroll
      for (int ni = 0; ni < 4; ++ni) {
        acc[mi][ni] = __builtin_amdgcn_wmma_f32_16x16x32_bf16(
            false, fa.v, false, fb[ni].v, (short)0, acc[mi][ni], false, false);
      }
    }
  }

  #pragma unroll
  for (int mi = 0; mi < 2; ++mi) {
    #pragma unroll
    for (int ni = 0; ni < 4; ++ni) {
      int col = blockN + wn * 64 + ni * 16 + l15;
      float bv = bias[col];
      int rbase = blockM + wm * 32 + mi * 16 + lhi * 8;
      #pragma unroll
      for (int i = 0; i < 8; ++i) {
        float v = acc[mi][ni][i] + bv;
        if (act) v = fmaxf(v, 0.0f);
        C[(size_t)(rbase + i) * Nn + col] = v;
      }
    }
  }
}

// 64x64-tile variant for the small, latency-critical recurrent GEMM (M=1024).
// 8 waves: 4 along M (16 rows each), 2 along N (32 cols each). Pipelined too.
__global__ __launch_bounds__(256) void gemm_small_wmma(
    const float* __restrict__ A, const float* __restrict__ B,
    const float* __restrict__ bias, float* __restrict__ C,
    int M, int Nn, int K, int act)
{
  __shared__ __align__(16) __bf16 Alds[64 * LDA_S];
  __shared__ __align__(16) __bf16 Blds[64 * LDA_S];

  const int tid  = threadIdx.x;
  const int lane = tid & 31;
  const int w    = tid >> 5;
  const int wm   = w & 3;            // row group: wm*16
  const int wn   = w >> 2;           // col group: wn*32
  const int blockM = blockIdx.y * 64;
  const int blockN = blockIdx.x * 64;
  const int l15 = lane & 15;
  const int lhi = lane >> 4;

  f32x8 acc[2];
  const f32x8 fzero = {};
  acc[0] = fzero; acc[1] = fzero;

  float4 ra[2], rb[2];
  auto load_tile = [&](int k0) {
    #pragma unroll
    for (int j = 0; j < 2; ++j) {
      int chunk = tid + j * 256;               // 512 float4 chunks each
      int r = chunk >> 3, kq = (chunk & 7) * 4;
      ra[j] = *(const float4*)(A + (size_t)(blockM + r) * K + (k0 + kq));
      int kk = chunk >> 4, nq = (chunk & 15) * 4;
      rb[j] = *(const float4*)(B + (size_t)(k0 + kk) * Nn + (blockN + nq));
    }
  };

  load_tile(0);
  for (int k0 = 0; k0 < K; k0 += BK) {
    __syncthreads();
    #pragma unroll
    for (int j = 0; j < 2; ++j) {
      int chunk = tid + j * 256;
      int r = chunk >> 3, kq = (chunk & 7) * 4;
      *(uint2*)(&Alds[r * LDA_S + kq]) = pack4bf(ra[j]);
      int kk = chunk >> 4, nq = (chunk & 15) * 4;
      Blds[(nq + 0) * LDA_S + kk] = f2bf(rb[j].x);
      Blds[(nq + 1) * LDA_S + kk] = f2bf(rb[j].y);
      Blds[(nq + 2) * LDA_S + kk] = f2bf(rb[j].z);
      Blds[(nq + 3) * LDA_S + kk] = f2bf(rb[j].w);
    }
    __syncthreads();
    if (k0 + BK < K) load_tile(k0 + BK);

    union FA { uint4 q[2]; bf16x16 v; } fa;
    const __bf16* ap = &Alds[(wm * 16 + l15) * LDA_S + lhi * 8];
    fa.q[0] = *(const uint4*)(ap);
    fa.q[1] = *(const uint4*)(ap + 16);
    #pragma unroll
    for (int ni = 0; ni < 2; ++ni) {
      union FB { uint4 q[2]; bf16x16 v; } fb;
      const __bf16* bp = &Blds[(wn * 32 + ni * 16 + l15) * LDA_S + lhi * 8];
      fb.q[0] = *(const uint4*)(bp);
      fb.q[1] = *(const uint4*)(bp + 16);
      acc[ni] = __builtin_amdgcn_wmma_f32_16x16x32_bf16(
          false, fa.v, false, fb.v, (short)0, acc[ni], false, false);
    }
  }

  #pragma unroll
  for (int ni = 0; ni < 2; ++ni) {
    int col = blockN + wn * 32 + ni * 16 + l15;
    float bv = bias[col];
    int rbase = blockM + wm * 16 + lhi * 8;
    #pragma unroll
    for (int i = 0; i < 8; ++i) {
      float v = acc[ni][i] + bv;
      if (act) v = fmaxf(v, 0.0f);
      C[(size_t)(rbase + i) * Nn + col] = v;
    }
  }
}

// ---------------- elementwise / scatter kernels ----------------
__global__ void zero_kernel(float4* p, size_t n4) {
  size_t i = (size_t)blockIdx.x * blockDim.x + threadIdx.x;
  size_t st = (size_t)gridDim.x * blockDim.x;
  float4 z = {0.f, 0.f, 0.f, 0.f};
  for (; i < n4; i += st) p[i] = z;
}
__global__ void copy_kernel(const float4* __restrict__ s, float4* __restrict__ d, size_t n4) {
  size_t i = (size_t)blockIdx.x * blockDim.x + threadIdx.x;
  size_t st = (size_t)gridDim.x * blockDim.x;
  for (; i < n4; i += st) d[i] = s[i];
}
// booster scatter: S += v, Mx = max(Mx, v). v >= 0 (post-ReLU) so uint max works.
__global__ void scatter_kernel(const float* __restrict__ hb, const int* __restrict__ row,
                               float* __restrict__ S, float* __restrict__ Mx) {
  int idx = blockIdx.x * 256 + threadIdx.x;
  if (idx >= NE * HD) return;
  int e = idx >> 8, c = idx & 255;
  int r = row[e];
  float v = hb[idx];
  atomicAdd(&S[(size_t)r * HD + c], v);
  atomicMax((unsigned int*)&Mx[(size_t)r * HD + c], __float_as_uint(v));
}
__global__ void combine_kernel(const float4* __restrict__ ha, const float4* __restrict__ S,
                               const float4* __restrict__ Mx, float* __restrict__ dst,
                               int dstStride) {
  int idx = blockIdx.x * 256 + threadIdx.x;           // over NA * 64
  if (idx >= NA * (HD / 4)) return;
  int n = idx >> 6, c4 = (idx & 63) * 4;
  float4 a = ha[idx], s = S[idx], m = Mx[idx];
  float4 r = {a.x + s.x * m.x, a.y + s.y * m.y, a.z + s.z * m.z, a.w + s.w * m.w};
  *(float4*)(dst + (size_t)n * dstStride + c4) = r;
}
__global__ void copy_strided_kernel(const float4* __restrict__ src, float* __restrict__ dst,
                                    int dstStride) {
  int idx = blockIdx.x * 256 + threadIdx.x;           // over NA * 64
  if (idx >= NA * (HD / 4)) return;
  int n = idx >> 6, c4 = (idx & 63) * 4;
  *(float4*)(dst + (size_t)n * dstStride + c4) = src[idx];
}
__global__ void edge_diff_kernel(const float* __restrict__ ha, const float* __restrict__ hb,
                                 const int* __restrict__ row, const int* __restrict__ col,
                                 float4* __restrict__ D) {
  int idx = blockIdx.x * 256 + threadIdx.x;           // over NE * 64
  if (idx >= NE * (HD / 4)) return;
  int e = idx >> 6, c4 = (idx & 63) * 4;
  float4 a = *(const float4*)(ha + (size_t)row[e] * HD + c4);
  float4 b = *(const float4*)(hb + (size_t)col[e] * HD + c4);
  float4 r = {a.x - b.x, a.y - b.y, a.z - b.z, a.w - b.w};
  D[idx] = r;
}
__global__ void bias_relu_kernel(const float4* __restrict__ h, const float4* __restrict__ b,
                                 float4* __restrict__ out) {
  int idx = blockIdx.x * 256 + threadIdx.x;           // over NA * 64
  if (idx >= NA * (HD / 4)) return;
  float4 v = h[idx], bb = b[idx & 63];
  float4 r = {fmaxf(v.x + bb.x, 0.f), fmaxf(v.y + bb.y, 0.f),
              fmaxf(v.z + bb.z, 0.f), fmaxf(v.w + bb.w, 0.f)};
  out[idx] = r;
}
__global__ void segmax_kernel(const float* __restrict__ h, float* __restrict__ h0) {
  int idx = blockIdx.x * 256 + threadIdx.x;
  if (idx >= NG * HD) return;
  int g = idx >> 8, c = idx & 255;
  float m = -INFINITY;
  for (int t = 0; t < LP; ++t) m = fmaxf(m, h[(size_t)(g * LP + t) * HD + c]);
  h0[idx] = m;
}
__global__ void transpose_kernel(const float* __restrict__ src, float* __restrict__ dst,
                                 int R, int C) {
  int idx = blockIdx.x * 256 + threadIdx.x;
  if (idx >= R * C) return;
  int r = idx / C, c = idx % C;
  dst[(size_t)c * R + r] = src[idx];
}
// GRU cell: gi precomputed per (graph,t); gh from recurrent GEMM; in-place h update.
__global__ void gru_cell_kernel(const float* __restrict__ GI, const float* __restrict__ GH,
                                float* __restrict__ hstate, float* __restrict__ CAT,
                                int t, int dsel) {
  int idx = blockIdx.x * 256 + threadIdx.x;
  if (idx >= NG * HD) return;
  int g = idx >> 8, c = idx & 255;
  size_t gi0 = (size_t)(g * LP + t) * (3 * HD);
  size_t gh0 = (size_t)g * (3 * HD);
  float i_r = GI[gi0 + c], i_z = GI[gi0 + HD + c], i_n = GI[gi0 + 2 * HD + c];
  float h_r = GH[gh0 + c], h_z = GH[gh0 + HD + c], h_n = GH[gh0 + 2 * HD + c];
  float hst = hstate[idx];
  float r = 1.0f / (1.0f + __expf(-(i_r + h_r)));
  float z = 1.0f / (1.0f + __expf(-(i_z + h_z)));
  float nn = tanhf(i_n + r * h_n);
  float hn = (1.0f - z) * nn + z * hst;
  hstate[idx] = hn;
  CAT[(size_t)(g * LP + t) * (2 * HD) + dsel * HD + c] = hn;
}

// ---------------- driver ----------------
extern "C" void kernel_launch(void* const* d_in, const int* in_sizes, int n_in,
                              void* d_out, int out_size, void* d_ws, size_t ws_size,
                              hipStream_t stream) {
  (void)in_sizes; (void)n_in; (void)out_size; (void)ws_size;
  const float* x        = (const float*)d_in[0];
  const float* edge_att = (const float*)d_in[1];
  const int*   eidx     = (const int*)d_in[2];
  const float* w_atom   = (const float*)d_in[4];
  const float* b_atom   = (const float*)d_in[5];
  const float* w_bond   = (const float*)d_in[6];
  const float* b_bond   = (const float*)d_in[7];
  const float* conv_w   = (const float*)d_in[8];
  const float* conv_b   = (const float*)d_in[9];
  const float* w_lin    = (const float*)d_in[10];
  const float* b_lin    = (const float*)d_in[11];
  const float* gru_bias = (const float*)d_in[12];
  const float* w_ih     = (const float*)d_in[13];
  const float* w_hh     = (const float*)d_in[14];
  const float* b_ih     = (const float*)d_in[15];
  const float* b_hh     = (const float*)d_in[16];
  const float* w_out    = (const float*)d_in[17];
  const float* b_out    = (const float*)d_in[18];
  const int* row = eidx;
  const int* col = eidx + NE;

  // workspace layout (floats)
  float* ws = (float*)d_ws;
  size_t o = 0;
  float* XPROJ = ws + o; o += (size_t)NA * HD;
  float* HATOM = ws + o; o += (size_t)NA * HD;
  float* HBOND = ws + o; o += (size_t)NE * HD;
  float* DBUF  = ws + o; o += (size_t)NE * HD;      // edge GEMM input; later CAT [NA,2H]
  float* SBUF  = ws + o; o += (size_t)NA * HD;      // SBUF,MBUF contiguous (one zero pass)
  float* MBUF  = ws + o; o += (size_t)NA * HD;
  float* CAT3  = ws + o; o += (size_t)NA * 3 * HD;  // concat input; later GI forward
  float* HH    = ws + o; o += (size_t)NA * HD;
  float* MSG   = ws + o; o += (size_t)NA * HD;
  float* GIB   = ws + o; o += (size_t)NA * 3 * HD;
  float* H0B   = ws + o; o += (size_t)NG * HD;
  float* HST   = ws + o; o += (size_t)NG * HD;
  float* GHB   = ws + o; o += (size_t)NG * 3 * HD;
  float* WIH0T = ws + o; o += (size_t)3 * HD * HD;
  float* WIH1T = ws + o; o += (size_t)3 * HD * HD;
  float* WHH0T = ws + o; o += (size_t)3 * HD * HD;
  float* WHH1T = ws + o; o += (size_t)3 * HD * HD;
  float* CAT   = DBUF;
  float* GIF   = CAT3;

  auto gemm = [&](const float* A, const float* B, const float* bias, float* C,
                  int M, int Nn, int K, int act) {
    dim3 grid(Nn / BN, M / BM);
    gemm_bf16_wmma<<<grid, 256, 0, stream>>>(A, B, bias, C, M, Nn, K, act);
  };
  auto gemm_s = [&](const float* A, const float* B, const float* bias, float* C,
                    int M, int Nn, int K, int act) {
    dim3 grid(Nn / 64, M / 64);
    gemm_small_wmma<<<grid, 256, 0, stream>>>(A, B, bias, C, M, Nn, K, act);
  };
  auto blocks = [](long n) { return (unsigned)((n + 255) / 256); };

  // 1) projections
  gemm(x, w_atom, b_atom, XPROJ, NA, HD, INC, 1);
  copy_kernel<<<2048, 256, 0, stream>>>((const float4*)XPROJ, (float4*)HATOM,
                                        (size_t)NA * HD / 4);
  gemm(edge_att, w_bond, b_bond, HBOND, NE, HD, EDIM, 1);

  // 2) three conv layers with additive booster
  for (int i = 0; i < 3; ++i) {
    zero_kernel<<<2048, 256, 0, stream>>>((float4*)SBUF, (size_t)2 * NA * HD / 4);
    scatter_kernel<<<blocks((long)NE * HD), 256, 0, stream>>>(HBOND, row, SBUF, MBUF);
    combine_kernel<<<blocks((long)NA * HD / 4), 256, 0, stream>>>(
        (const float4*)HATOM, (const float4*)SBUF, (const float4*)MBUF, HATOM, HD);
    edge_diff_kernel<<<blocks((long)NE * HD / 4), 256, 0, stream>>>(
        HATOM, HBOND, row, col, (float4*)DBUF);
    gemm(DBUF, conv_w + (size_t)i * HD * HD, conv_b + (size_t)i * HD, HBOND, NE, HD, HD, 1);
  }
  // 3) final booster -> aggr goes into CAT3[:,0:H]; build concat
  zero_kernel<<<2048, 256, 0, stream>>>((float4*)SBUF, (size_t)2 * NA * HD / 4);
  scatter_kernel<<<blocks((long)NE * HD), 256, 0, stream>>>(HBOND, row, SBUF, MBUF);
  combine_kernel<<<blocks((long)NA * HD / 4), 256, 0, stream>>>(
      (const float4*)HATOM, (const float4*)SBUF, (const float4*)MBUF, CAT3, 3 * HD);
  copy_strided_kernel<<<blocks((long)NA * HD / 4), 256, 0, stream>>>(
      (const float4*)HATOM, CAT3 + HD, 3 * HD);
  copy_strided_kernel<<<blocks((long)NA * HD / 4), 256, 0, stream>>>(
      (const float4*)XPROJ, CAT3 + 2 * HD, 3 * HD);
  gemm(CAT3, w_lin, b_lin, HH, NA, HD, 3 * HD, 0);

  // 4) GRU prep
  bias_relu_kernel<<<blocks((long)NA * HD / 4), 256, 0, stream>>>(
      (const float4*)HH, (const float4*)gru_bias, (float4*)MSG);
  segmax_kernel<<<blocks((long)NG * HD), 256, 0, stream>>>(HH, H0B);
  transpose_kernel<<<blocks(3 * HD * HD), 256, 0, stream>>>(w_ih, WIH0T, 3 * HD, HD);
  transpose_kernel<<<blocks(3 * HD * HD), 256, 0, stream>>>(w_ih + (size_t)3 * HD * HD, WIH1T, 3 * HD, HD);
  transpose_kernel<<<blocks(3 * HD * HD), 256, 0, stream>>>(w_hh, WHH0T, 3 * HD, HD);
  transpose_kernel<<<blocks(3 * HD * HD), 256, 0, stream>>>(w_hh + (size_t)3 * HD * HD, WHH1T, 3 * HD, HD);
  // hoisted input gates: one big GEMM per direction
  gemm(MSG, WIH0T, b_ih, GIF, NA, 3 * HD, HD, 0);
  gemm(MSG, WIH1T, b_ih + 3 * HD, GIB, NA, 3 * HD, HD, 0);

  // 5) bidirectional GRU: only the small recurrent GEMM is sequential; use the
  //    64x64-tile kernel for 4x more blocks per step (192 vs 48).
  for (int d = 0; d < 2; ++d) {
    copy_kernel<<<1024, 256, 0, stream>>>((const float4*)H0B, (float4*)HST,
                                          (size_t)NG * HD / 4);
    const float* GI = d ? GIB : GIF;
    const float* WT = d ? WHH1T : WHH0T;
    const float* bh = b_hh + (size_t)d * 3 * HD;
    for (int s = 0; s < LP; ++s) {
      int t = d ? (LP - 1 - s) : s;
      gemm_s(HST, WT, bh, GHB, NG, 3 * HD, HD, 0);
      gru_cell_kernel<<<blocks((long)NG * HD), 256, 0, stream>>>(GI, GHB, HST, CAT, t, d);
    }
  }

  // 6) output head: [NA, 2H] @ [2H, H] + relu
  gemm(CAT, w_out, b_out, (float*)d_out, NA, HD, 2 * HD, 1);
}
